// get_model_12541304504438
// MI455X (gfx1250) — matrix operand order repaired and checked by reading, not compiled, not run on hardware
//
#include <hip/hip_runtime.h>
#include <hip/hip_bf16.h>
#include <stdint.h>

typedef __bf16 bf16_t;
typedef __attribute__((ext_vector_type(16))) __bf16 v16bf;
typedef __attribute__((ext_vector_type(8)))  __bf16 v8bf;
typedef __attribute__((ext_vector_type(8)))  float  v8f;

union V16U { v16bf v; struct { v8bf lo, hi; } p; };

constexpr int kB   = 16;
constexpr int kN   = 8192;
constexpr int kNP1 = 512;
constexpr int kNS1 = 32;
constexpr int kNP2 = 128;
constexpr int kNS2 = 64;
constexpr int kP3  = 128;
constexpr int kM1  = kB * kNP1 * kNS1;   // 262144
constexpr int kM2  = kB * kNP2 * kNS2;   // 131072
constexpr int kM3  = kB * kP3;           // 2048

// ---------------------------------------------------------------------------
// Transpose (B,6,N) -> p_xyz (B,N,3), norm (B,N,3)
// ---------------------------------------------------------------------------
__global__ void transpose_kernel(const float* __restrict__ xyz,
                                 float* __restrict__ pxyz,
                                 float* __restrict__ nrm) {
  int idx = blockIdx.x * blockDim.x + threadIdx.x;
  if (idx >= kB * kN) return;
  int b = idx / kN, n = idx % kN;
  const float* src = xyz + (size_t)b * 6 * kN + n;
  float* dp = pxyz + (size_t)idx * 3;
  float* dn = nrm  + (size_t)idx * 3;
  dp[0] = src[0 * kN]; dp[1] = src[1 * kN]; dp[2] = src[2 * kN];
  dn[0] = src[3 * kN]; dn[1] = src[4 * kN]; dn[2] = src[5 * kN];
}

// ---------------------------------------------------------------------------
// Farthest point sampling: one workgroup per batch, dist[] resident in LDS,
// wave32 shuffle argmax + cross-wave LDS reduce each of npoint rounds.
// ---------------------------------------------------------------------------
__global__ __launch_bounds__(1024)
void fps_kernel(const float* __restrict__ pts, int n, int npoint,
                int* __restrict__ fidx) {
  __shared__ float dist[8192];
  __shared__ float rv[32];
  __shared__ int   ri[32];
  __shared__ int   sfar;
  int b = blockIdx.x;
  const float* P = pts + (size_t)b * n * 3;
  int tid = threadIdx.x;
  int lane = tid & 31, wid = tid >> 5;
  int nw = blockDim.x >> 5;
  for (int j = tid; j < n; j += blockDim.x) dist[j] = 1e10f;
  if (tid == 0) sfar = 0;
  __syncthreads();
  for (int i = 0; i < npoint; ++i) {
    int far = sfar;
    if (tid == 0) fidx[b * npoint + i] = far;
    float fx = P[far * 3 + 0], fy = P[far * 3 + 1], fz = P[far * 3 + 2];
    float bv = -1.f; int bi = 0x7fffffff;
    for (int j = tid; j < n; j += blockDim.x) {
      float dx = P[j * 3 + 0] - fx;
      float dy = P[j * 3 + 1] - fy;
      float dz = P[j * 3 + 2] - fz;
      float d = dx * dx + dy * dy + dz * dz;
      float dm = fminf(dist[j], d);
      dist[j] = dm;
      if (dm > bv || (dm == bv && j < bi)) { bv = dm; bi = j; }
    }
    #pragma unroll
    for (int off = 16; off > 0; off >>= 1) {
      float ov = __shfl_down(bv, off, 32);
      int   oi = __shfl_down(bi, off, 32);
      if (ov > bv || (ov == bv && oi < bi)) { bv = ov; bi = oi; }
    }
    if (lane == 0) { rv[wid] = bv; ri[wid] = bi; }
    __syncthreads();
    if (wid == 0) {
      float v2 = (lane < nw) ? rv[lane] : -1.f;
      int   i2 = (lane < nw) ? ri[lane] : 0x7fffffff;
      #pragma unroll
      for (int off = 16; off > 0; off >>= 1) {
        float ov = __shfl_down(v2, off, 32);
        int   oi = __shfl_down(i2, off, 32);
        if (ov > v2 || (ov == v2 && oi < i2)) { v2 = ov; i2 = oi; }
      }
      if (lane == 0) sfar = i2;
    }
    __syncthreads();
  }
}

__global__ void gather_xyz_kernel(const float* __restrict__ pts,
                                  const int* __restrict__ fidx,
                                  float* __restrict__ nxyz,
                                  int n, int npoint) {
  int idx = blockIdx.x * blockDim.x + threadIdx.x;
  if (idx >= kB * npoint) return;
  int b = idx / npoint;
  int f = fidx[idx];
  const float* s = pts + ((size_t)b * n + f) * 3;
  float* d = nxyz + (size_t)idx * 3;
  d[0] = s[0]; d[1] = s[1]; d[2] = s[2];
}

// ---------------------------------------------------------------------------
// Ball query: one wave per query; ballot picks the first `nsample` in-radius
// indices in increasing order (== sorted semantics of the reference).
// ---------------------------------------------------------------------------
__global__ void ballquery_kernel(const float* __restrict__ pts,
                                 const float* __restrict__ centers,
                                 int* __restrict__ gidx,
                                 int n, int npoint, int nsample, float r2) {
  int wpb = blockDim.x >> 5;
  int q = blockIdx.x * wpb + (threadIdx.x >> 5);
  int lane = threadIdx.x & 31;
  int b = q / npoint;
  const float* P = pts + (size_t)b * n * 3;
  float cx = centers[q * 3 + 0];
  float cy = centers[q * 3 + 1];
  float cz = centers[q * 3 + 2];
  int* out = gidx + (size_t)q * nsample;
  int cnt = 0, first = 0;
  for (int base = 0; base < n && cnt < nsample; base += 32) {
    int j = base + lane;
    float dx = P[j * 3 + 0] - cx;
    float dy = P[j * 3 + 1] - cy;
    float dz = P[j * 3 + 2] - cz;
    bool pred = (dx * dx + dy * dy + dz * dz) <= r2;
    unsigned mask = __builtin_amdgcn_ballot_w32(pred);
    if (cnt == 0 && mask != 0u) first = base + __builtin_ctz(mask);
    int prefix = __popc(mask & ((1u << lane) - 1u));
    int slot = cnt + prefix;
    if (pred && slot < nsample) out[slot] = j;
    cnt += __popc(mask);
  }
  if (cnt < nsample)
    for (int s = cnt + lane; s < nsample; s += 32) out[s] = first;
}

// ---------------------------------------------------------------------------
// Grouped feature builders (write bf16 activation rows, K zero-padded to 32x)
// ---------------------------------------------------------------------------
__global__ void build_x1_kernel(const float* __restrict__ pxyz,
                                const float* __restrict__ nrm,
                                const float* __restrict__ nxyz1,
                                const int* __restrict__ gidx1,
                                bf16_t* __restrict__ X) {
  int row = blockIdx.x * blockDim.x + threadIdx.x;
  if (row >= kM1) return;
  int bq = row / kNS1;
  int b = bq / kNP1;
  int g = gidx1[row];
  const float* pp = pxyz + ((size_t)b * kN + g) * 3;
  const float* nn = nrm  + ((size_t)b * kN + g) * 3;
  const float* c  = nxyz1 + (size_t)bq * 3;
  bf16_t* o = X + (size_t)row * 32;
  o[0] = (bf16_t)(pp[0] - c[0]);
  o[1] = (bf16_t)(pp[1] - c[1]);
  o[2] = (bf16_t)(pp[2] - c[2]);
  o[3] = (bf16_t)nn[0]; o[4] = (bf16_t)nn[1]; o[5] = (bf16_t)nn[2];
  #pragma unroll
  for (int k = 6; k < 32; ++k) o[k] = (bf16_t)0.f;
}

__global__ void build_x2_kernel(const float* __restrict__ nxyz1,
                                const float* __restrict__ l1f,
                                const float* __restrict__ nxyz2,
                                const int* __restrict__ gidx2,
                                bf16_t* __restrict__ X) {
  int row = blockIdx.x * blockDim.x + threadIdx.x;
  if (row >= kM2) return;
  int bq = row / kNS2;
  int b = bq / kNP2;
  int g = gidx2[row];
  const float* ps = nxyz1 + ((size_t)b * kNP1 + g) * 3;
  const float* c  = nxyz2 + (size_t)bq * 3;
  const float* f  = l1f + ((size_t)b * kNP1 + g) * 128;
  bf16_t* o = X + (size_t)row * 160;
  o[0] = (bf16_t)(ps[0] - c[0]);
  o[1] = (bf16_t)(ps[1] - c[1]);
  o[2] = (bf16_t)(ps[2] - c[2]);
  for (int k = 0; k < 128; ++k) o[3 + k] = (bf16_t)f[k];
  #pragma unroll
  for (int k = 131; k < 160; ++k) o[k] = (bf16_t)0.f;
}

__global__ void build_x3_kernel(const float* __restrict__ nxyz2,
                                const float* __restrict__ l2f,
                                bf16_t* __restrict__ X) {
  int row = blockIdx.x * blockDim.x + threadIdx.x;
  if (row >= kM3) return;
  const float* c = nxyz2 + (size_t)row * 3;
  const float* f = l2f + (size_t)row * 256;
  bf16_t* o = X + (size_t)row * 288;
  o[0] = (bf16_t)c[0]; o[1] = (bf16_t)c[1]; o[2] = (bf16_t)c[2];
  for (int k = 0; k < 256; ++k) o[3 + k] = (bf16_t)f[k];
  #pragma unroll
  for (int k = 259; k < 288; ++k) o[k] = (bf16_t)0.f;
}

// ---------------------------------------------------------------------------
// Weight prep: pack W (Cout,Cin) f32 into WMMA B-fragment order, bf16,
// K zero-padded. Fragment layout: lane%16<->n, (lane/16, elem)<->k in table order.
// ---------------------------------------------------------------------------
__global__ void prep_w_kernel(const float* __restrict__ W,
                              bf16_t* __restrict__ Wpk,
                              int Cin, int Cout, int Kpad) {
  int pck = blockIdx.x * blockDim.x + threadIdx.x;
  if (pck >= Cout * Kpad) return;
  int KT = Kpad >> 5;
  int e = pck & 15;
  int l = (pck >> 4) & 31;
  int rest = pck >> 9;
  int kti = rest % KT;
  int nt  = rest / KT;
  int row = nt * 16 + (l & 15);
  int k = kti * 32 + ((e & 8) << 1) + (l >> 4) * 8 + (e & 7);
  float v = (k < Cin) ? W[(size_t)row * Cin + k] : 0.f;
  Wpk[pck] = (bf16_t)v;
}

// Fold bias + batchnorm into per-channel scale/shift. g==nullptr => plain bias.
__global__ void prep_ss_kernel(const float* __restrict__ bb,
                               const float* __restrict__ g,
                               const float* __restrict__ bt,
                               const float* __restrict__ m,
                               const float* __restrict__ v,
                               float* __restrict__ scale,
                               float* __restrict__ shift, int Cout) {
  int n = blockIdx.x * blockDim.x + threadIdx.x;
  if (n >= Cout) return;
  if (g) {
    float inv = rsqrtf(v[n] + 1e-5f);
    float s = g[n] * inv;
    scale[n] = s;
    shift[n] = (bb[n] - m[n]) * s + bt[n];
  } else {
    scale[n] = 1.f;
    shift[n] = bb[n];
  }
}

__device__ inline void epi_store(bf16_t* outB, float* outF, int relu, int Cout,
                                 int m, int n, float acc, float s, float t) {
  float y = acc * s + t;
  if (relu) y = fmaxf(y, 0.f);
  if (outB) outB[(size_t)m * Cout + n] = (bf16_t)y;
  if (outF) outF[(size_t)m * Cout + n] = y;
}

__device__ inline v16bf load_a(const bf16_t* p) {
  V16U u;
  u.p.lo = *(const v8bf*)(p);
  u.p.hi = *(const v8bf*)(p + 16);
  return u.v;
}
__device__ inline v16bf load_b(const bf16_t* p) {
  V16U u;
  u.p.lo = *(const v8bf*)(p);
  u.p.hi = *(const v8bf*)(p + 8);
  return u.v;
}

// ---------------------------------------------------------------------------
// Register-blocked GEMM: each wave computes a 32x32 output block (2x2 tiles),
// 4 v_wmma per 4 fragment loads -> 2x the flops/byte of the 1x1 version.
// Requires M % 32 == 0 && Cout % 32 == 0.
// ---------------------------------------------------------------------------
__global__ __launch_bounds__(256)
void gemm2x2_bn_relu_kernel(const bf16_t* __restrict__ X,
                            const bf16_t* __restrict__ Wpk,
                            const float* __restrict__ scale,
                            const float* __restrict__ shift,
                            int M, int Kpad, int Cout, int relu,
                            bf16_t* __restrict__ outB,
                            float* __restrict__ outF) {
  int wg = blockIdx.x * (blockDim.x >> 5) + (threadIdx.x >> 5);
  int NT2 = Cout >> 5;
  int tiles = (M >> 5) * NT2;
  if (wg >= tiles) return;               // wave-uniform; EXEC stays all-ones
  int mt = (wg / NT2) * 2, nt = (wg % NT2) * 2;
  int l = threadIdx.x & 31;
  int half = l >> 4, lm = l & 15;
  int KT = Kpad >> 5;
  const bf16_t* rowA0 = X + (size_t)(mt * 16 + lm) * Kpad + half * 8;
  const bf16_t* rowA1 = rowA0 + (size_t)16 * Kpad;
  const bf16_t* pB0 = Wpk + ((size_t)nt * KT * 32 + l) * 16;
  const bf16_t* pB1 = pB0 + (size_t)KT * 32 * 16;
  v8f c00 = {0.f,0.f,0.f,0.f,0.f,0.f,0.f,0.f};
  v8f c01 = c00, c10 = c00, c11 = c00;
  for (int kt = 0; kt < KT; ++kt) {
    v16bf a0 = load_a(rowA0);
    v16bf a1 = load_a(rowA1);
    v16bf b0 = load_b(pB0);
    v16bf b1 = load_b(pB1);
    c00 = __builtin_amdgcn_wmma_f32_16x16x32_bf16(false, a0, false, b0, (short)0, c00, false, false);
    c01 = __builtin_amdgcn_wmma_f32_16x16x32_bf16(false, a0, false, b1, (short)0, c01, false, false);
    c10 = __builtin_amdgcn_wmma_f32_16x16x32_bf16(false, a1, false, b0, (short)0, c10, false, false);
    c11 = __builtin_amdgcn_wmma_f32_16x16x32_bf16(false, a1, false, b1, (short)0, c11, false, false);
    rowA0 += 32; rowA1 += 32;
    pB0 += 32 * 16; pB1 += 32 * 16;
  }
  int n0 = nt * 16 + lm, n1 = n0 + 16;
  float s0 = scale[n0], t0 = shift[n0];
  float s1 = scale[n1], t1 = shift[n1];
  #pragma unroll
  for (int j = 0; j < 8; ++j) {
    int m0 = mt * 16 + j + half * 8;     // C/D layout: VGPR j -> row j+8*half
    int m1 = m0 + 16;
    epi_store(outB, outF, relu, Cout, m0, n0, c00[j], s0, t0);
    epi_store(outB, outF, relu, Cout, m0, n1, c01[j], s1, t1);
    epi_store(outB, outF, relu, Cout, m1, n0, c10[j], s0, t0);
    epi_store(outB, outF, relu, Cout, m1, n1, c11[j], s1, t1);
  }
}

// 1x1-tile GEMM for the tiny head layers (M == 16).
__global__ __launch_bounds__(256)
void gemm_bn_relu_kernel(const bf16_t* __restrict__ X,
                         const bf16_t* __restrict__ Wpk,
                         const float* __restrict__ scale,
                         const float* __restrict__ shift,
                         int M, int Kpad, int Cout, int relu,
                         bf16_t* __restrict__ outB,
                         float* __restrict__ outF) {
  int wg = blockIdx.x * (blockDim.x >> 5) + (threadIdx.x >> 5);
  int Ntiles = Cout >> 4;
  int tiles = (M >> 4) * Ntiles;
  if (wg >= tiles) return;
  int mt = wg / Ntiles, nt = wg % Ntiles;
  int l = threadIdx.x & 31;
  int half = l >> 4, lm = l & 15;
  int KT = Kpad >> 5;
  const bf16_t* rowA = X + (size_t)(mt * 16 + lm) * Kpad + half * 8;
  const bf16_t* pB = Wpk + ((size_t)nt * KT * 32 + l) * 16;
  v8f c = {0.f,0.f,0.f,0.f,0.f,0.f,0.f,0.f};
  for (int kt = 0; kt < KT; ++kt) {
    v16bf a = load_a(rowA);
    v16bf b = load_b(pB);
    c = __builtin_amdgcn_wmma_f32_16x16x32_bf16(false, a, false, b, (short)0, c, false, false);
    rowA += 32;
    pB += 32 * 16;
  }
  int n = nt * 16 + lm;
  float s = scale[n], t = shift[n];
  #pragma unroll
  for (int j = 0; j < 8; ++j)
    epi_store(outB, outF, relu, Cout, mt * 16 + j + half * 8, n, c[j], s, t);
}

// Max over the sample axis: (G,S,C) bf16 -> (G,C)
__global__ void maxreduce_kernel(const bf16_t* __restrict__ in,
                                 float* __restrict__ outF,
                                 bf16_t* __restrict__ outB,
                                 int G, int S, int C) {
  int idx = blockIdx.x * blockDim.x + threadIdx.x;
  if (idx >= G * C) return;
  int g = idx / C, ch = idx % C;
  const bf16_t* p = in + (size_t)g * S * C + ch;
  float m = -1e30f;
  for (int s = 0; s < S; ++s) m = fmaxf(m, (float)p[(size_t)s * C]);
  if (outF) outF[idx] = m;
  if (outB) outB[idx] = (bf16_t)m;
}

// ---------------------------------------------------------------------------
extern "C" void kernel_launch(void* const* d_in, const int* in_sizes, int n_in,
                              void* d_out, int out_size, void* d_ws, size_t ws_size,
                              hipStream_t stream) {
  (void)in_sizes; (void)n_in; (void)out_size; (void)ws_size;
  auto F = [&](int i) { return (const float*)d_in[i]; };
  const float* xyz = F(0);

  // ---- bump allocator over d_ws ----
  char* wp = (char*)d_ws;
  auto alloc = [&](size_t bytes) -> void* {
    void* r = (void*)wp;
    wp += (bytes + 255) & ~((size_t)255);
    return r;
  };
  float*  PXYZ  = (float*)alloc((size_t)kB * kN * 3 * 4);
  float*  NORM  = (float*)alloc((size_t)kB * kN * 3 * 4);
  int*    FIDX1 = (int*)alloc((size_t)kB * kNP1 * 4);
  float*  NXYZ1 = (float*)alloc((size_t)kB * kNP1 * 3 * 4);
  int*    GIDX1 = (int*)alloc((size_t)kB * kNP1 * kNS1 * 4);
  float*  L1F   = (float*)alloc((size_t)kB * kNP1 * 128 * 4);
  int*    FIDX2 = (int*)alloc((size_t)kB * kNP2 * 4);
  float*  NXYZ2 = (float*)alloc((size_t)kB * kNP2 * 3 * 4);
  int*    GIDX2 = (int*)alloc((size_t)kB * kNP2 * kNS2 * 4);
  float*  L2F   = (float*)alloc((size_t)kB * kNP2 * 256 * 4);
  bf16_t* HBUF  = (bf16_t*)alloc((size_t)16 * 1024 * 2);
  bf16_t* HH1   = (bf16_t*)alloc((size_t)16 * 512 * 2);
  bf16_t* HH2   = (bf16_t*)alloc((size_t)16 * 256 * 2);
  bf16_t* WPK   = (bf16_t*)alloc((size_t)1024 * 512 * 2);  // max Cout*Kpad
  float*  SC    = (float*)alloc(1024 * 4);
  float*  SH    = (float*)alloc(1024 * 4);
  bf16_t* BUFA  = (bf16_t*)alloc((size_t)20971520 * 2);    // max of X1,H1b,X2,H2b,X3,H3b
  bf16_t* BUFB  = (bf16_t*)alloc((size_t)33554432 * 2);    // max of H1a,H1c,H2a,H2c,H3a,H3c

  float* houtF = (float*)d_out;             // h: 16x1024
  float* zoutF = (float*)d_out + 16 * 1024; // z: 16x128

  // prep + GEMM for one linear(+BN)(+ReLU) layer
  auto run_layer = [&](const bf16_t* Xin, int M, int Cin, int Kpad, int Cout,
                       const float* W, const float* bb, const float* g,
                       const float* bt, const float* m, const float* v,
                       bool relu, bf16_t* outB, float* outF) {
    int tw = Cout * Kpad;
    prep_w_kernel<<<(tw + 255) / 256, 256, 0, stream>>>(W, WPK, Cin, Cout, Kpad);
    prep_ss_kernel<<<(Cout + 255) / 256, 256, 0, stream>>>(bb, g, bt, m, v, SC, SH, Cout);
    if ((M % 32) == 0 && (Cout % 32) == 0) {
      int tiles = (M / 32) * (Cout / 32);
      gemm2x2_bn_relu_kernel<<<(tiles + 7) / 8, 256, 0, stream>>>(
          Xin, WPK, SC, SH, M, Kpad, Cout, relu ? 1 : 0, outB, outF);
    } else {
      int tiles = (M / 16) * (Cout / 16);
      gemm_bn_relu_kernel<<<(tiles + 7) / 8, 256, 0, stream>>>(
          Xin, WPK, SC, SH, M, Kpad, Cout, relu ? 1 : 0, outB, outF);
    }
  };
  auto bn_layer = [&](const bf16_t* Xin, int M, int Cin, int Kpad, int Cout,
                      int base, bf16_t* outB) {
    run_layer(Xin, M, Cin, Kpad, Cout, F(base + 0), F(base + 1), F(base + 2),
              F(base + 3), F(base + 4), F(base + 5), true, outB, nullptr);
  };

  // ---- stage 0: transpose ----
  transpose_kernel<<<(kB * kN + 255) / 256, 256, 0, stream>>>(xyz, PXYZ, NORM);

  // ---- SA1 ----
  fps_kernel<<<kB, 1024, 0, stream>>>(PXYZ, kN, kNP1, FIDX1);
  gather_xyz_kernel<<<(kB * kNP1 + 255) / 256, 256, 0, stream>>>(PXYZ, FIDX1, NXYZ1, kN, kNP1);
  ballquery_kernel<<<(kB * kNP1) / 8, 256, 0, stream>>>(PXYZ, NXYZ1, GIDX1, kN, kNP1, kNS1, 0.04f);
  bf16_t* X1 = BUFA;
  build_x1_kernel<<<(kM1 + 255) / 256, 256, 0, stream>>>(PXYZ, NORM, NXYZ1, GIDX1, X1);
  bf16_t* H1a = BUFB; bf16_t* H1b = BUFA; bf16_t* H1c = BUFB;
  bn_layer(X1,  kM1,  6,  32,  64, 1,  H1a);
  bn_layer(H1a, kM1, 64,  64,  64, 7,  H1b);
  bn_layer(H1b, kM1, 64,  64, 128, 13, H1c);
  maxreduce_kernel<<<(kB * kNP1 * 128 + 255) / 256, 256, 0, stream>>>(
      H1c, L1F, nullptr, kB * kNP1, kNS1, 128);

  // ---- SA2 ----
  fps_kernel<<<kB, 1024, 0, stream>>>(NXYZ1, kNP1, kNP2, FIDX2);
  gather_xyz_kernel<<<(kB * kNP2 + 255) / 256, 256, 0, stream>>>(NXYZ1, FIDX2, NXYZ2, kNP1, kNP2);
  ballquery_kernel<<<(kB * kNP2) / 8, 256, 0, stream>>>(NXYZ1, NXYZ2, GIDX2, kNP1, kNP2, kNS2, 0.16f);
  bf16_t* X2 = BUFA;
  build_x2_kernel<<<(kM2 + 255) / 256, 256, 0, stream>>>(NXYZ1, L1F, NXYZ2, GIDX2, X2);
  bf16_t* H2a = BUFB; bf16_t* H2b = BUFA; bf16_t* H2c = BUFB;
  bn_layer(X2,  kM2, 131, 160, 128, 19, H2a);
  bn_layer(H2a, kM2, 128, 128, 128, 25, H2b);
  bn_layer(H2b, kM2, 128, 128, 256, 31, H2c);
  maxreduce_kernel<<<(kB * kNP2 * 256 + 255) / 256, 256, 0, stream>>>(
      H2c, L2F, nullptr, kB * kNP2, kNS2, 256);

  // ---- SA3 (global) ----
  bf16_t* X3 = BUFA;
  build_x3_kernel<<<(kM3 + 255) / 256, 256, 0, stream>>>(NXYZ2, L2F, X3);
  bf16_t* H3a = BUFB; bf16_t* H3b = BUFA; bf16_t* H3c = BUFB;
  bn_layer(X3,  kM3, 259, 288,  256, 37, H3a);
  bn_layer(H3a, kM3, 256, 256,  512, 43, H3b);
  bn_layer(H3b, kM3, 512, 512, 1024, 49, H3c);
  maxreduce_kernel<<<(kB * 1024 + 255) / 256, 256, 0, stream>>>(
      H3c, houtF, HBUF, kB, kP3, 1024);   // h -> d_out and bf16 copy for head

  // ---- head ----
  run_layer(HBUF, 16, 1024, 1024, 512, F(55), F(56), F(57), F(58), F(59), F(60),
            true, HH1, nullptr);
  run_layer(HH1, 16, 512, 512, 256, F(61), F(62), F(63), F(64), F(65), F(66),
            true, HH2, nullptr);
  run_layer(HH2, 16, 256, 256, 128, F(67), F(68), nullptr, nullptr, nullptr,
            nullptr, false, nullptr, zoutF);  // z -> d_out tail
}
